// UKT_41944650612763
// MI455X (gfx1250) — compile-verified, matrix operand order checked by reading, not run on hardware
//
#include <hip/hip_runtime.h>
#include <math.h>
#include <stdint.h>

// ---------------------------------------------------------------- constants
#define BB   4
#define TT   1024
#define DD   512
#define HH   8
#define DFF_ 2048
#define DKK  64
#define KCH  128            // K-chunk staged in LDS (bf16), padded stride 136

typedef __attribute__((ext_vector_type(16))) __bf16 v16bf;
typedef __attribute__((ext_vector_type(8)))  __bf16 v8bf;
typedef __attribute__((ext_vector_type(8)))  float  v8f;
typedef __attribute__((ext_vector_type(4)))  unsigned int v4u;
typedef __attribute__((ext_vector_type(8)))  int v8i;
typedef __attribute__((ext_vector_type(4)))  int v4i;

#define HD __device__ __forceinline__

#if __has_builtin(__builtin_amdgcn_tensor_load_to_lds) && \
    __has_builtin(__builtin_amdgcn_s_wait_tensorcnt)
#define USE_TDM 1
#else
#define USE_TDM 0
#endif

// ---------------------------------------------------------------- helpers
HD float fsqrt_fast(float x) { return __builtin_amdgcn_sqrtf(x); }

HD v8f wmma_bf16(v16bf a, v16bf b, v8f c) {
  return __builtin_amdgcn_wmma_f32_16x16x32_bf16(
      false, a, false, b, (short)0, c, false, false);
}

// Fragment layout (wave32, 16x16x32 bf16): lane = output index (A:M, B:N),
// halves 0..7 = K in [sel*8, sel*8+8), halves 8..15 = K in [16+sel*8, ...).
// Caller passes p already offset by (row*ld + k + sel*8).
HD v16bf ldfrag_f32(const float* p) {
  v8f lo = *(const v8f*)p;
  v8f hi = *(const v8f*)(p + 16);
  v16bf r;
#pragma unroll
  for (int i = 0; i < 8; ++i) { r[i] = (__bf16)lo[i]; r[i + 8] = (__bf16)hi[i]; }
  return r;
}
HD v16bf ldfrag_f32_sqrt(const float* p) {  // sqrt(clip(x,1e-24)) fused
  v8f lo = *(const v8f*)p;
  v8f hi = *(const v8f*)(p + 16);
  v16bf r;
#pragma unroll
  for (int i = 0; i < 8; ++i) {
    r[i]     = (__bf16)fsqrt_fast(fmaxf(lo[i], 1e-24f));
    r[i + 8] = (__bf16)fsqrt_fast(fmaxf(hi[i], 1e-24f));
  }
  return r;
}
HD v16bf ldfrag_bf(const __bf16* p) {
  v8bf lo = *(const v8bf*)p;
  v8bf hi = *(const v8bf*)(p + 16);
  v16bf r;
#pragma unroll
  for (int i = 0; i < 8; ++i) { r[i] = lo[i]; r[i + 8] = hi[i]; }
  return r;
}

HD float wsum(float v) {
#pragma unroll
  for (int o = 16; o > 0; o >>= 1) v += __shfl_xor(v, o, 32);
  return v;
}
HD float wmaxr(float v) {
#pragma unroll
  for (int o = 16; o > 0; o >>= 1) v = fmaxf(v, __shfl_xor(v, o, 32));
  return v;
}

#if USE_TDM
// Tensor Data Mover: load a 32-row x 128-col bf16 tile into LDS with 16B
// padding every 256B (-> LDS row stride 136 halves). D# per CDNA5 ISA ch.8.
HD void tdm_load_tile_bf16(const __bf16* gsrc, unsigned lds_off, int row_stride) {
  unsigned long long ga = (unsigned long long)(uintptr_t)gsrc;
  v4u g0;
  g0[0] = 1u;                                    // count=1 (valid user D#)
  g0[1] = lds_off;                               // lds_addr (bytes)
  g0[2] = (unsigned)ga;                          // global_addr[31:0]
  g0[3] = (unsigned)((ga >> 32) & 0x01FFFFFFull) | (2u << 30);  // [56:32]|type=2
  v8i g1;
  // data_size=2B(code1), pad_enable, pad_interval=64dw(code5), pad_amount=4dw(code3)
  g1[0] = (int)((1u << 16) | (1u << 20) | (5u << 22) | (3u << 25));
  g1[1] = (int)0xFFFF0000u;                      // tensor_dim0=0x7FFFFFFF lo16
  g1[2] = (int)0xFFFF7FFFu;                      // dim0 hi | tensor_dim1 lo16
  g1[3] = (int)(0x00007FFFu | (128u << 16));     // dim1 hi | tile_dim0=128
  g1[4] = 32;                                    // tile_dim1=32, tile_dim2=0
  g1[5] = row_stride;                            // tensor_dim0_stride (elems)
  g1[6] = 0;
  g1[7] = 0;
  v4i z4 = {0, 0, 0, 0};
  v8i z8 = {0, 0, 0, 0, 0, 0, 0, 0};
  // clang-23 lane: 6-arg form (g0, g1, g2, g3, extra, cpol)
  __builtin_amdgcn_tensor_load_to_lds(g0, g1, z4, z4, z8, 0);
}
#endif

// Stage a shared 32 x 128 bf16 panel (global, row stride ld) into LDS
// (row stride 136). Caller wraps with __syncthreads() before/after.
HD void stage_panel32_128(const __bf16* g, int ld, __bf16* lds0) {
#if USE_TDM
  if (threadIdx.x < 32) {  // wave 0 drives the DMA; others wait at barrier
    tdm_load_tile_bf16(g, (unsigned)(uintptr_t)lds0, ld);
    __builtin_amdgcn_s_wait_tensorcnt(0);
  }
#else
  for (int c = threadIdx.x; c < 512; c += 256) {
    int rr = c >> 4, cc = (c & 15) * 8;
    *(v8bf*)(lds0 + rr * 136 + cc) = *(const v8bf*)(g + (size_t)rr * ld + cc);
  }
#endif
}

// ---------------------------------------------------------------- kernels

// Weight transpose + bf16 convert: W[K,N] f32 -> WT[N,K] bf16
__global__ void wt_kernel(const float* __restrict__ W, __bf16* __restrict__ WT,
                          int K, int N) {
  int i = blockIdx.x * 256 + threadIdx.x;
  if (i >= K * N) return;
  int n = i % N, k = i / N;
  WT[(size_t)n * K + k] = (__bf16)W[i];
}

// Generic GEMM: C[M,N] = act(A[M,K] (f32) @ WT[N,K]^T (bf16) + bias), f32 out.
// Wave tile 16(M) x 32(N); 8 waves/block share the 32xK B panel via LDS/TDM.
template <int ACT>
__global__ void gemm_kernel(const float* __restrict__ A,
                            const __bf16* __restrict__ BT,
                            const float* __restrict__ bias,
                            float* __restrict__ C, int M, int N, int K) {
  __shared__ __attribute__((aligned(16))) __bf16 lb[32 * 136];
  int wave = threadIdx.x >> 5, lane = threadIdx.x & 31;
  int m0 = (blockIdx.y * 8 + wave) * 16;
  int n0 = blockIdx.x * 32;
  int r = lane & 15, sel = lane >> 4;
  const float* arow = A + (size_t)(m0 + r) * K + sel * 8;
  v8f acc0 = {}, acc1 = {};
  for (int kc = 0; kc < K; kc += KCH) {
    __syncthreads();
    stage_panel32_128(BT + (size_t)n0 * K + kc, K, lb);
    __syncthreads();
    if (kc + KCH < K) __builtin_prefetch(arow + kc + KCH, 0, 3);
#pragma unroll
    for (int k = 0; k < KCH; k += 32) {
      v16bf a  = ldfrag_f32(arow + kc + k);
      v16bf b0 = ldfrag_bf(lb + r * 136 + k + sel * 8);
      v16bf b1 = ldfrag_bf(lb + (16 + r) * 136 + k + sel * 8);
      acc0 = wmma_bf16(a, b0, acc0);
      acc1 = wmma_bf16(a, b1, acc1);
    }
  }
  int n = lane & 15;
  float bi0 = bias[n0 + n], bi1 = bias[n0 + 16 + n];
#pragma unroll
  for (int j = 0; j < 8; ++j) {
    int m = m0 + j + sel * 8;
    float v0 = acc0[j] + bi0, v1 = acc1[j] + bi1;
    if (ACT == 1) { v0 = fmaxf(v0, 0.f); v1 = fmaxf(v1, 0.f); }
    C[(size_t)m * N + n0 + n]      = v0;
    C[(size_t)m * N + n0 + 16 + n] = v1;
  }
}

// Per (b,h,t): sum_d pm^2 and sum_d pc over the head slice.
__global__ void rowstats_kernel(const float* __restrict__ Pm,
                                const float* __restrict__ Pc,
                                float* __restrict__ M2, float* __restrict__ SC) {
  int i = blockIdx.x * 256 + threadIdx.x;  // (b,h,t) flat
  int t = i % TT, h = (i / TT) % HH, b = i / (TT * HH);
  const float* pm = Pm + ((size_t)(b * TT + t)) * DD + h * DKK;
  const float* pc = Pc + ((size_t)(b * TT + t)) * DD + h * DKK;
  float m2 = 0.f, sc = 0.f;
#pragma unroll 8
  for (int d = 0; d < DKK; ++d) { m2 += pm[d] * pm[d]; sc += pc[d]; }
  M2[i] = m2; SC[i] = sc;
}

// Raw Wasserstein scores: S[t,s] = (2*(qm.km + sq.sk) - (|qm|^2+|km|^2+Sqc+Skc))/8.
// K-side panel (km rows + sqrt(kc) rows, shared by all 8 waves) staged in LDS
// with the sqrt computed ONCE per block. Upper-triangle tiles culled.
__global__ void scores_kernel(const float* __restrict__ Qm, const float* __restrict__ Qc,
                              const float* __restrict__ Km, const float* __restrict__ Kc,
                              const float* __restrict__ m2q, const float* __restrict__ scq,
                              const float* __restrict__ m2k, const float* __restrict__ sck,
                              __bf16* __restrict__ S) {
  __shared__ __attribute__((aligned(16))) __bf16 lkm[32 * 72];
  __shared__ __attribute__((aligned(16))) __bf16 lks[32 * 72];
  int wave = threadIdx.x >> 5, lane = threadIdx.x & 31;
  int bh = blockIdx.z, b = bh / HH, h = bh % HH;
  int t0 = (blockIdx.y * 8 + wave) * 16;
  int s0 = blockIdx.x * 32;
  {  // cooperative staging: 32 rows x 64 cols, one 8-elem chunk per thread
    int c = threadIdx.x;
    int rr = c >> 3, cc = (c & 7) * 8;
    const float* pm = Km + (size_t)(b * TT + s0 + rr) * DD + h * DKK + cc;
    const float* pc = Kc + (size_t)(b * TT + s0 + rr) * DD + h * DKK + cc;
    v8f vm_ = *(const v8f*)pm;
    v8f vc_ = *(const v8f*)pc;
    v8bf om, oc;
#pragma unroll
    for (int j = 0; j < 8; ++j) {
      om[j] = (__bf16)vm_[j];
      oc[j] = (__bf16)fsqrt_fast(fmaxf(vc_[j], 1e-24f));
    }
    *(v8bf*)(lkm + rr * 72 + cc) = om;
    *(v8bf*)(lks + rr * 72 + cc) = oc;
  }
  __syncthreads();
  if (s0 >= t0 + 15) return;  // no s < t in this tile
  int r = lane & 15, sel = lane >> 4;
  const size_t rowA = (size_t)(b * TT + t0 + r) * DD + h * DKK + sel * 8;
  v8f acc0 = {}, acc1 = {};
#pragma unroll
  for (int k = 0; k < DKK; k += 32) {
    v16bf am  = ldfrag_f32(Qm + rowA + k);
    v16bf ac  = ldfrag_f32_sqrt(Qc + rowA + k);
    v16bf bm0 = ldfrag_bf(lkm + r * 72 + k + sel * 8);
    v16bf bc0 = ldfrag_bf(lks + r * 72 + k + sel * 8);
    v16bf bm1 = ldfrag_bf(lkm + (16 + r) * 72 + k + sel * 8);
    v16bf bc1 = ldfrag_bf(lks + (16 + r) * 72 + k + sel * 8);
    acc0 = wmma_bf16(am, bm0, acc0);
    acc0 = wmma_bf16(ac, bc0, acc0);
    acc1 = wmma_bf16(am, bm1, acc1);
    acc1 = wmma_bf16(ac, bc1, acc1);
  }
  int n = lane & 15;
  float bt0 = m2k[(size_t)bh * TT + s0 + n]      + sck[(size_t)bh * TT + s0 + n];
  float bt1 = m2k[(size_t)bh * TT + s0 + 16 + n] + sck[(size_t)bh * TT + s0 + 16 + n];
#pragma unroll
  for (int j = 0; j < 8; ++j) {
    int t = t0 + j + sel * 8;
    float at = m2q[(size_t)bh * TT + t] + scq[(size_t)bh * TT + t];
    float v0 = (2.f * acc0[j] - (at + bt0)) * 0.125f;
    float v1 = (2.f * acc1[j] - (at + bt1)) * 0.125f;
    size_t o = ((size_t)bh * TT + t) * TT + s0 + n;
    S[o] = (__bf16)v0; S[o + 16] = (__bf16)v1;
  }
}

// One wave per row: softmax -> cumsum decay -> second softmax; in-place bf16.
__global__ void rowsm_kernel(__bf16* __restrict__ S,
                             const float* __restrict__ gammas) {
  int wave = threadIdx.x >> 5, lane = threadIdx.x & 31;
  int row = blockIdx.x * 8 + wave;  // (b,h,t) flat
  int t = row % TT, h = (row / TT) % HH;
  __bf16* srow = S + (size_t)row * TT;
  int sbase = lane * 32;
  if (t == 0) {  // zero_pad: row 0 all zeros
    v8bf z = {};
    v8bf* wp = (v8bf*)(srow + sbase);
    wp[0] = z; wp[1] = z; wp[2] = z; wp[3] = z;
    return;
  }
  float x[32];
  {
    const v8bf* vp = (const v8bf*)(srow + sbase);
#pragma unroll
    for (int c = 0; c < 4; ++c) {
      v8bf v = vp[c];
#pragma unroll
      for (int j = 0; j < 8; ++j) x[c * 8 + j] = (float)v[j];
    }
  }
  float gamma = -log1pf(__expf(gammas[h]));  // -softplus
  // first softmax over s < t
  float mx = -3.4e38f;
#pragma unroll
  for (int i = 0; i < 32; ++i) if (sbase + i < t) mx = fmaxf(mx, x[i]);
  mx = wmaxr(mx);
  float sv[32];
  float suml = 0.f;
#pragma unroll
  for (int i = 0; i < 32; ++i) {
    sv[i] = (sbase + i < t) ? __expf(x[i] - mx) : 0.f;
    suml += sv[i];
  }
  float tot = wsum(suml);
  float inv = 1.f / tot;
  // wave-wide inclusive scan of per-lane probability mass
  float lane_total = suml * inv;
  float scan = lane_total;
#pragma unroll
  for (int d = 1; d < 32; d <<= 1) {
    float tv = __shfl_up(scan, d, 32);
    if (lane >= d) scan += tv;
  }
  float excl = scan - lane_total;
  float tot2 = __shfl(scan, 31, 32);  // disttot
  float run = excl;
#pragma unroll
  for (int i = 0; i < 32; ++i) {
    int s = sbase + i;
    run += sv[i] * inv;  // inclusive cumsum of s_
    float ds = fsqrt_fast(fmaxf((tot2 - run) * fabsf((float)(t - s)), 0.f));
    float te = fminf(fmaxf(__expf(ds * gamma), 1e-5f), 1e5f);
    x[i] = (s < t) ? x[i] * te : -3.4e38f;
  }
  float mx2 = -3.4e38f;
#pragma unroll
  for (int i = 0; i < 32; ++i) mx2 = fmaxf(mx2, x[i]);
  mx2 = wmaxr(mx2);
  float sum2 = 0.f;
#pragma unroll
  for (int i = 0; i < 32; ++i) {
    sv[i] = (sbase + i < t) ? __expf(x[i] - mx2) : 0.f;
    sum2 += sv[i];
  }
  sum2 = wsum(sum2);
  float ip = 1.f / sum2;
  {
    v8bf* wp = (v8bf*)(srow + sbase);
#pragma unroll
    for (int c = 0; c < 4; ++c) {
      v8bf v;
#pragma unroll
      for (int j = 0; j < 8; ++j) v[j] = (__bf16)(sv[c * 8 + j] * ip);
      wp[c] = v;
    }
  }
}

// V transpose: (B,T,D) f32 -> (B,H,DK,T) bf16
__global__ void vtrans_kernel(const float* __restrict__ V, __bf16* __restrict__ VT) {
  size_t i = (size_t)blockIdx.x * 256 + threadIdx.x;
  int s = (int)(i % TT);
  size_t rest = i / TT;
  int d = (int)(rest % DKK); rest /= DKK;
  int h = (int)(rest % HH);
  int b = (int)(rest / HH);
  VT[i] = (__bf16)V[(size_t)(b * TT + s) * DD + h * DKK + d];
}

// PV: Om = P @ Vm, Oc = P^2 @ Vc per (b,h); writes (B,T,D) layout. Shared V
// panels staged in LDS (TDM when available).
__global__ void pv_kernel(const __bf16* __restrict__ S,
                          const __bf16* __restrict__ VmT,
                          const __bf16* __restrict__ VcT,
                          float* __restrict__ Om, float* __restrict__ Oc) {
  __shared__ __attribute__((aligned(16))) __bf16 lvm[32 * 136];
  __shared__ __attribute__((aligned(16))) __bf16 lvc[32 * 136];
  int wave = threadIdx.x >> 5, lane = threadIdx.x & 31;
  int bh = blockIdx.z, b = bh / HH, h = bh % HH;
  int t0 = (blockIdx.y * 8 + wave) * 16;
  int n0 = blockIdx.x * 32;
  int r = lane & 15, sel = lane >> 4;
  const __bf16* arow = S + ((size_t)bh * TT + t0 + r) * TT + sel * 8;
  const __bf16* vmb  = VmT + ((size_t)bh * DKK + n0) * TT;
  const __bf16* vcb  = VcT + ((size_t)bh * DKK + n0) * TT;
  v8f am0 = {}, am1 = {}, ac0 = {}, ac1 = {};
  for (int kc = 0; kc < TT; kc += KCH) {
    __syncthreads();
    stage_panel32_128(vmb + kc, TT, lvm);
    stage_panel32_128(vcb + kc, TT, lvc);
    __syncthreads();
    if (kc + KCH < TT) __builtin_prefetch(arow + kc + KCH, 0, 3);
#pragma unroll
    for (int k = 0; k < KCH; k += 32) {
      v16bf p = ldfrag_bf(arow + kc + k);
      v16bf p2;
#pragma unroll
      for (int i = 0; i < 16; ++i) {
        float f = (float)p[i];
        p2[i] = (__bf16)(f * f);
      }
      v16bf vm0 = ldfrag_bf(lvm + r * 136 + k + sel * 8);
      v16bf vm1 = ldfrag_bf(lvm + (16 + r) * 136 + k + sel * 8);
      v16bf vc0 = ldfrag_bf(lvc + r * 136 + k + sel * 8);
      v16bf vc1 = ldfrag_bf(lvc + (16 + r) * 136 + k + sel * 8);
      am0 = wmma_bf16(p, vm0, am0);
      am1 = wmma_bf16(p, vm1, am1);
      ac0 = wmma_bf16(p2, vc0, ac0);
      ac1 = wmma_bf16(p2, vc1, ac1);
    }
  }
  int n = lane & 15;
#pragma unroll
  for (int j = 0; j < 8; ++j) {
    int t = t0 + j + sel * 8;
    size_t o = ((size_t)(b * TT + t)) * DD + h * DKK + n0 + n;
    Om[o]      = am0[j];
    Om[o + 16] = am1[j];
    Oc[o]      = ac0[j];
    Oc[o + 16] = ac1[j];
  }
}

// Residual + LN (mean) and residual + elu + 1 + LN (cov). Wave per row.
__global__ void ln1_kernel(const float* __restrict__ q_mean, const float* __restrict__ Am,
                           const float* __restrict__ q_cov,  const float* __restrict__ Ac,
                           const float* __restrict__ w, const float* __restrict__ bv,
                           float* __restrict__ Ym, float* __restrict__ Yc) {
  int wave = threadIdx.x >> 5, lane = threadIdx.x & 31;
  int row = blockIdx.x * 8 + wave;  // B*T rows
  size_t base = (size_t)row * DD;
  float xm[16], xc[16];
#pragma unroll
  for (int i = 0; i < 16; ++i) {
    int d = lane * 16 + i;
    xm[i] = q_mean[base + d] + Am[base + d];
    float c = q_cov[base + d] + Ac[base + d];
    xc[i] = ((c > 0.f) ? c : expm1f(c)) + 1.f;
  }
  float s = 0.f, s2 = 0.f;
#pragma unroll
  for (int i = 0; i < 16; ++i) { s += xm[i]; s2 += xm[i] * xm[i]; }
  s = wsum(s); s2 = wsum(s2);
  float mean = s * (1.f / DD), var = s2 * (1.f / DD) - mean * mean;
  float rs = rsqrtf(var + 1e-5f);
#pragma unroll
  for (int i = 0; i < 16; ++i) {
    int d = lane * 16 + i;
    Ym[base + d] = (xm[i] - mean) * rs * w[d] + bv[d];
  }
  s = 0.f; s2 = 0.f;
#pragma unroll
  for (int i = 0; i < 16; ++i) { s += xc[i]; s2 += xc[i] * xc[i]; }
  s = wsum(s); s2 = wsum(s2);
  mean = s * (1.f / DD); var = s2 * (1.f / DD) - mean * mean;
  rs = rsqrtf(var + 1e-5f);
#pragma unroll
  for (int i = 0; i < 16; ++i) {
    int d = lane * 16 + i;
    Yc[base + d] = (xc[i] - mean) * rs * w[d] + bv[d];
  }
}

// Final LN: out[0] = LN(ffm), out[1] = LN(elu(ffc)+1). Wave per row.
__global__ void ln2_kernel(const float* __restrict__ Fm, const float* __restrict__ Fc,
                           const float* __restrict__ w, const float* __restrict__ bv,
                           float* __restrict__ out) {
  int wave = threadIdx.x >> 5, lane = threadIdx.x & 31;
  int row = blockIdx.x * 8 + wave;
  size_t base = (size_t)row * DD;
  const size_t half = (size_t)BB * TT * DD;
  float xm[16], xc[16];
#pragma unroll
  for (int i = 0; i < 16; ++i) {
    int d = lane * 16 + i;
    xm[i] = Fm[base + d];
    float c = Fc[base + d];
    xc[i] = ((c > 0.f) ? c : expm1f(c)) + 1.f;
  }
  float s = 0.f, s2 = 0.f;
#pragma unroll
  for (int i = 0; i < 16; ++i) { s += xm[i]; s2 += xm[i] * xm[i]; }
  s = wsum(s); s2 = wsum(s2);
  float mean = s * (1.f / DD), var = s2 * (1.f / DD) - mean * mean;
  float rs = rsqrtf(var + 1e-5f);
#pragma unroll
  for (int i = 0; i < 16; ++i) {
    int d = lane * 16 + i;
    out[base + d] = (xm[i] - mean) * rs * w[d] + bv[d];
  }
  s = 0.f; s2 = 0.f;
#pragma unroll
  for (int i = 0; i < 16; ++i) { s += xc[i]; s2 += xc[i] * xc[i]; }
  s = wsum(s); s2 = wsum(s2);
  mean = s * (1.f / DD); var = s2 * (1.f / DD) - mean * mean;
  rs = rsqrtf(var + 1e-5f);
#pragma unroll
  for (int i = 0; i < 16; ++i) {
    int d = lane * 16 + i;
    out[half + base + d] = (xc[i] - mean) * rs * w[d] + bv[d];
  }
}

// ---------------------------------------------------------------- launch
extern "C" void kernel_launch(void* const* d_in, const int* in_sizes, int n_in,
                              void* d_out, int out_size, void* d_ws, size_t ws_size,
                              hipStream_t stream) {
  (void)in_sizes; (void)n_in; (void)out_size; (void)ws_size;
  const float* q_mean = (const float*)d_in[0];
  const float* q_cov  = (const float*)d_in[1];
  const float* k_mean = (const float*)d_in[2];
  const float* k_cov  = (const float*)d_in[3];
  const float* v_mean = (const float*)d_in[4];
  const float* v_cov  = (const float*)d_in[5];
  const float* Wk_mean = (const float*)d_in[6];
  const float* bk_mean = (const float*)d_in[7];
  const float* Wk_cov  = (const float*)d_in[8];
  const float* bk_cov  = (const float*)d_in[9];
  const float* Wv_mean = (const float*)d_in[10];
  const float* bv_mean = (const float*)d_in[11];
  const float* Wv_cov  = (const float*)d_in[12];
  const float* bv_cov  = (const float*)d_in[13];
  const float* Wom = (const float*)d_in[14];
  const float* bom = (const float*)d_in[15];
  const float* Woc = (const float*)d_in[16];
  const float* boc = (const float*)d_in[17];
  const float* gammas = (const float*)d_in[18];
  const float* ln1w = (const float*)d_in[19];
  const float* ln1b = (const float*)d_in[20];
  const float* ln2w = (const float*)d_in[21];
  const float* ln2b = (const float*)d_in[22];
  const float* W1m = (const float*)d_in[23];
  const float* b1m = (const float*)d_in[24];
  const float* W1c = (const float*)d_in[25];
  const float* b1c = (const float*)d_in[26];
  const float* W2m = (const float*)d_in[27];
  const float* b2m = (const float*)d_in[28];
  const float* W2c = (const float*)d_in[29];
  const float* b2c = (const float*)d_in[30];
  float* out = (float*)d_out;

  const size_t BT   = (size_t)BB * TT;
  const size_t BTD  = BT * DD;
  const size_t BHT  = (size_t)BB * HH * TT;
  const size_t BHTT = (size_t)BB * HH * TT * TT;

  char* wsp = (char*)d_ws;
  size_t off = 0;
  auto alloc = [&](size_t bytes) -> void* {
    void* p = wsp + off;
    off += (bytes + 255) & ~(size_t)255;
    return p;
  };

  __bf16* wt_km = (__bf16*)alloc((size_t)DD * DD * 2);
  __bf16* wt_kc = (__bf16*)alloc((size_t)DD * DD * 2);
  __bf16* wt_vm = (__bf16*)alloc((size_t)DD * DD * 2);
  __bf16* wt_vc = (__bf16*)alloc((size_t)DD * DD * 2);
  __bf16* wt_om = (__bf16*)alloc((size_t)DD * DD * 2);
  __bf16* wt_oc = (__bf16*)alloc((size_t)DD * DD * 2);
  __bf16* wt_1m = (__bf16*)alloc((size_t)DD * DFF_ * 2);
  __bf16* wt_1c = (__bf16*)alloc((size_t)DD * DFF_ * 2);
  __bf16* wt_2m = (__bf16*)alloc((size_t)DFF_ * DD * 2);
  __bf16* wt_2c = (__bf16*)alloc((size_t)DFF_ * DD * 2);
  float* qm = (float*)alloc(BTD * 4);
  float* qc = (float*)alloc(BTD * 4);
  float* km = (float*)alloc(BTD * 4);
  float* kc = (float*)alloc(BTD * 4);
  float* vm = (float*)alloc(BTD * 4);
  float* vc = (float*)alloc(BTD * 4);
  float* m2q = (float*)alloc(BHT * 4);
  float* scq = (float*)alloc(BHT * 4);
  float* m2k = (float*)alloc(BHT * 4);
  float* sck = (float*)alloc(BHT * 4);
  __bf16* vmT = (__bf16*)alloc(BTD * 2);
  __bf16* vcT = (__bf16*)alloc(BTD * 2);
  __bf16* S   = (__bf16*)alloc(BHTT * 2);
  // stage-based reuse (sequential stream makes this safe):
  float* Om = qm;            // PV mean out (qm f32 dead after scores)
  float* Oc = qc;
  float* projm = km;         // attn out-proj (km f32 dead after scores)
  float* projc = kc;
  float* qm1 = vm;           // post-LN1 (vm f32 dead after vtrans)
  float* qc1 = vc;
  float* hid = (float*)S;    // FFN hidden (S dead after PV)
  float* ffm = qm;           // FFN out (Om consumed by out-proj)
  float* ffc = qc;

  // 1) weight transpose+bf16
  wt_kernel<<<(DD * DD) / 256, 256, 0, stream>>>(Wk_mean, wt_km, DD, DD);
  wt_kernel<<<(DD * DD) / 256, 256, 0, stream>>>(Wk_cov,  wt_kc, DD, DD);
  wt_kernel<<<(DD * DD) / 256, 256, 0, stream>>>(Wv_mean, wt_vm, DD, DD);
  wt_kernel<<<(DD * DD) / 256, 256, 0, stream>>>(Wv_cov,  wt_vc, DD, DD);
  wt_kernel<<<(DD * DD) / 256, 256, 0, stream>>>(Wom,     wt_om, DD, DD);
  wt_kernel<<<(DD * DD) / 256, 256, 0, stream>>>(Woc,     wt_oc, DD, DD);
  wt_kernel<<<(DD * DFF_) / 256, 256, 0, stream>>>(W1m, wt_1m, DD, DFF_);
  wt_kernel<<<(DD * DFF_) / 256, 256, 0, stream>>>(W1c, wt_1c, DD, DFF_);
  wt_kernel<<<(DFF_ * DD) / 256, 256, 0, stream>>>(W2m, wt_2m, DFF_, DD);
  wt_kernel<<<(DFF_ * DD) / 256, 256, 0, stream>>>(W2c, wt_2c, DFF_, DD);

  // 2) projections (kq_same: q/k share Wk)
  dim3 gp(DD / 32, BT / 128);
  gemm_kernel<0><<<gp, 256, 0, stream>>>(q_mean, wt_km, bk_mean, qm, BT, DD, DD);
  gemm_kernel<0><<<gp, 256, 0, stream>>>(q_cov,  wt_kc, bk_cov,  qc, BT, DD, DD);
  gemm_kernel<0><<<gp, 256, 0, stream>>>(k_mean, wt_km, bk_mean, km, BT, DD, DD);
  gemm_kernel<0><<<gp, 256, 0, stream>>>(k_cov,  wt_kc, bk_cov,  kc, BT, DD, DD);
  gemm_kernel<0><<<gp, 256, 0, stream>>>(v_mean, wt_vm, bv_mean, vm, BT, DD, DD);
  gemm_kernel<0><<<gp, 256, 0, stream>>>(v_cov,  wt_vc, bv_cov,  vc, BT, DD, DD);

  // 3) per-row norm/sum terms
  rowstats_kernel<<<(unsigned)(BHT / 256), 256, 0, stream>>>(qm, qc, m2q, scq);
  rowstats_kernel<<<(unsigned)(BHT / 256), 256, 0, stream>>>(km, kc, m2k, sck);

  // 4) raw scores (WMMA, lower-triangular tiles only)
  scores_kernel<<<dim3(TT / 32, TT / 128, BB * HH), 256, 0, stream>>>(
      qm, qc, km, kc, m2q, scq, m2k, sck, S);

  // 5) softmax -> distance decay -> softmax (in-place, bf16 probs)
  rowsm_kernel<<<(unsigned)(BHT / 8), 256, 0, stream>>>(S, gammas);

  // 6) transpose V for WMMA B operand
  vtrans_kernel<<<(unsigned)(BTD / 256), 256, 0, stream>>>(vm, vmT);
  vtrans_kernel<<<(unsigned)(BTD / 256), 256, 0, stream>>>(vc, vcT);

  // 7) PV (p and p^2 against vm/vc)
  pv_kernel<<<dim3(DKK / 32, TT / 128, BB * HH), 256, 0, stream>>>(S, vmT, vcT, Om, Oc);

  // 8) output projections
  gemm_kernel<0><<<gp, 256, 0, stream>>>(Om, wt_om, bom, projm, BT, DD, DD);
  gemm_kernel<0><<<gp, 256, 0, stream>>>(Oc, wt_oc, boc, projc, BT, DD, DD);

  // 9) residual + LN1
  ln1_kernel<<<(unsigned)(BT / 8), 256, 0, stream>>>(q_mean, projm, q_cov, projc,
                                                     ln1w, ln1b, qm1, qc1);

  // 10) FFN (mean then cov, hidden buffer reused)
  dim3 gf1(DFF_ / 32, BT / 128), gf2(DD / 32, BT / 128);
  gemm_kernel<1><<<gf1, 256, 0, stream>>>(qm1, wt_1m, b1m, hid, BT, DFF_, DD);
  gemm_kernel<0><<<gf2, 256, 0, stream>>>(hid, wt_2m, b2m, ffm, BT, DD, DFF_);
  gemm_kernel<1><<<gf1, 256, 0, stream>>>(qc1, wt_1c, b1c, hid, BT, DFF_, DD);
  gemm_kernel<0><<<gf2, 256, 0, stream>>>(hid, wt_2c, b2c, ffc, BT, DD, DFF_);

  // 11) final LN -> stacked output
  ln2_kernel<<<(unsigned)(BT / 8), 256, 0, stream>>>(ffm, ffc, ln2w, ln2b, out);
}
